// PNALayer_13365938226036
// MI455X (gfx1250) — matrix operand order repaired and checked by reading, not compiled, not run on hardware
//
#include <hip/hip_runtime.h>
#include <hip/hip_bf16.h>

typedef __attribute__((ext_vector_type(16))) _Float16 v16h;
typedef __attribute__((ext_vector_type(8)))  _Float16 v8h;
typedef __attribute__((ext_vector_type(4)))  _Float16 v4h;
typedef __attribute__((ext_vector_type(2)))  _Float16 v2h;
typedef __attribute__((ext_vector_type(8)))  float    v8f;
typedef __attribute__((ext_vector_type(4)))  float    v4f;

#define NN 8192
#define EE 131072
#define TT 4
#define DIN 128
#define AVG_LOG 2.772588722239781f

// ---------- order-preserving float<->uint for atomic max/min ----------
__device__ __forceinline__ unsigned fenc(float x) {
    unsigned u = __float_as_uint(x);
    return (u >> 31) ? ~u : (u | 0x80000000u);
}
__device__ __forceinline__ float fdec(unsigned u) {
    return (u >> 31) ? __uint_as_float(u & 0x7FFFFFFFu) : __uint_as_float(~u);
}

// ---------- WMMA fragment loaders (contiguous k = 16*(lane>>4) + s, same for A and B) ----------
// A: 16xK tile, row-major f16, strideH halves per row, k0 pre-applied by caller.
// lane<16: row=lane, halves [0..15]=K[0..15]
// lane>=16: row=lane-16, halves [0..15]=K[16..31]
__device__ __forceinline__ v16h load_a_frag(const _Float16* base, int strideH, int lane) {
    int row = lane & 15;
    int kb  = (lane >> 4) << 4;                 // 0 or 16
    const _Float16* p = base + row * strideH + kb;
    v8h lo = *(const v8h*)(p);
    v8h hi = *(const v8h*)(p + 8);
    return __builtin_shufflevector(lo, hi, 0,1,2,3,4,5,6,7,8,9,10,11,12,13,14,15);
}
// B: 32x16 tile from transposed weights Bt[n][k] (strideH halves per n-row).
// lane holds column n=lane&15, K-half (lane>>4): 16 contiguous halves (one 32B load).
__device__ __forceinline__ v16h load_b_frag(const _Float16* bt, int strideH, int lane) {
    int n  = lane & 15;
    int kb = (lane >> 4) << 4;                  // 0 or 16
    return *(const v16h*)(bt + n * strideH + kb);
}

// ---------- K0: convert+transpose weights  W[b][K][Nc] f32 -> Wt[b][Nc][K] f16 ----------
__global__ void k_w2h(const float* __restrict__ W, _Float16* __restrict__ Wt, int K, int Nc) {
    long total = (long)K * Nc;
    long i = (long)blockIdx.x * blockDim.x + threadIdx.x;
    if (i >= total) return;
    long b = blockIdx.y;
    long k = i % K;
    long n = i / K;
    Wt[b * total + n * K + k] = (_Float16)W[b * total + k * Nc + n];
}

// ---------- K_init: clear accumulators every call (deterministic) ----------
__global__ void k_init(float* Ssum, float* Ssq, unsigned* Smax, unsigned* Smin, unsigned* deg) {
    long i = (long)blockIdx.x * blockDim.x + threadIdx.x;
    if (i < (long)NN * TT * DIN) {
        Ssum[i] = 0.f; Ssq[i] = 0.f;
        Smax[i] = 0u;             // below enc of any real value
        Smin[i] = 0xFFFFFFFFu;    // above enc of any real value
    }
    if (i < NN) deg[i] = 0u;
}

// ---------- K_deg ----------
__global__ void k_deg(const int* __restrict__ dst, unsigned* __restrict__ deg) {
    int e = blockIdx.x * blockDim.x + threadIdx.x;
    if (e < EE) atomicAdd(&deg[dst[e]], 1u);
}

// ---------- K2: pretrans (z = [hs|hd|e] @ preW + b per tower) + scatter-aggregate ----------
// block = 128 thr (4 waves), 16 edges; wave t does tower t.  K = 288 = 9*32.
__global__ void k_pretrans(const float* __restrict__ h, const float* __restrict__ e,
                           const float* __restrict__ preB,
                           const int* __restrict__ src, const int* __restrict__ dst,
                           const _Float16* __restrict__ preWt,
                           float* Ssum, float* Ssq, unsigned* Smax, unsigned* Smin) {
    __shared__ _Float16 zs[TT][16][296];        // 288 padded to 296 halves (16B rows)
    __shared__ int dsts[16];
    const int tid = threadIdx.x;
    const long ebase = (long)blockIdx.x * 16;

    if (tid < 16) dsts[tid] = dst[ebase + tid];
    // gather h[src] -> zs[t][ed][0..127]   (float4 -> packed 4x f16)
    for (int i = tid; i < 16 * 128; i += 128) {
        int ed = i >> 7, c4 = (i & 127) << 2;           // 0..508 step 4
        v4f v = *(const v4f*)(h + (long)src[ebase + ed] * 512 + c4);
        v4h x = { (_Float16)v.x, (_Float16)v.y, (_Float16)v.z, (_Float16)v.w };
        *(v4h*)&zs[c4 >> 7][ed][c4 & 127] = x;
    }
    // gather h[dst] -> zs[t][ed][128..255]
    for (int i = tid; i < 16 * 128; i += 128) {
        int ed = i >> 7, c4 = (i & 127) << 2;
        v4f v = *(const v4f*)(h + (long)dst[ebase + ed] * 512 + c4);
        v4h x = { (_Float16)v.x, (_Float16)v.y, (_Float16)v.z, (_Float16)v.w };
        *(v4h*)&zs[c4 >> 7][ed][128 + (c4 & 127)] = x;
    }
    // edge features broadcast -> zs[t][ed][256..287]
    for (int i = tid; i < 16 * 8; i += 128) {
        int ed = i >> 3, c4 = (i & 7) << 2;
        v4f v = *(const v4f*)(e + (ebase + ed) * 32 + c4);
        v4h x = { (_Float16)v.x, (_Float16)v.y, (_Float16)v.z, (_Float16)v.w };
        *(v4h*)&zs[0][ed][256 + c4] = x;
        *(v4h*)&zs[1][ed][256 + c4] = x;
        *(v4h*)&zs[2][ed][256 + c4] = x;
        *(v4h*)&zs[3][ed][256 + c4] = x;
    }
    __syncthreads();

    const int t = tid >> 5, lane = tid & 31;
    v8f acc[8];
    #pragma unroll
    for (int i = 0; i < 8; ++i)
        #pragma unroll
        for (int j = 0; j < 8; ++j) acc[i][j] = 0.f;

    const _Float16* A0 = &zs[t][0][0];
    const _Float16* Bt0 = preWt + (long)t * 128 * 288;
    for (int kk = 0; kk < 9; ++kk) {
        int k0 = kk * 32;
        v16h a = load_a_frag(A0 + k0, 296, lane);
        v16h bfr[8];
        #pragma unroll
        for (int nt = 0; nt < 8; ++nt)
            bfr[nt] = load_b_frag(Bt0 + (long)nt * 16 * 288 + k0, 288, lane);
        #pragma unroll
        for (int nt = 0; nt < 8; ++nt)
            acc[nt] = __builtin_amdgcn_wmma_f32_16x16x32_f16(false, a, false, bfr[nt],
                                                             (short)0, acc[nt], false, false);
    }
    // scatter m into segment accumulators (D layout: m = v + 8*(lane>=16), n = lane&15)
    const int n15 = lane & 15;
    const int mOff = (lane >> 4) * 8;
    #pragma unroll
    for (int nt = 0; nt < 8; ++nt) {
        int col = nt * 16 + n15;
        float bias = preB[t * 128 + col];
        #pragma unroll
        for (int v = 0; v < 8; ++v) {
            float val = acc[nt][v] + bias;
            long off = (long)dsts[mOff + v] * 512 + t * 128 + col;
            atomicAdd(&Ssum[off], val);
            atomicAdd(&Ssq[off], val * val);
            atomicMax(&Smax[off], fenc(val));
            atomicMin(&Smin[off], fenc(val));
        }
    }
}

// ---------- K3a: build ht[N][T][1664] f16 = [h | agg | agg*s1 | agg*s2] ----------
// 2-wide: component boundaries are multiples of 128, so an even pair never straddles.
__global__ void k_build_ht(const float* __restrict__ h, const float* __restrict__ Ssum,
                           const float* __restrict__ Ssq, const unsigned* __restrict__ Smax,
                           const unsigned* __restrict__ Smin, const unsigned* __restrict__ degA,
                           _Float16* __restrict__ ht) {
    const int ntid = blockIdx.x;                 // n*4 + t
    const int n = ntid >> 2, t = ntid & 3;
    const float deg = (float)degA[n];
    const bool  has = deg > 0.f;
    const float degc = fmaxf(deg, 1.f);
    const float invd = 1.f / degc;
    const float dlog = logf(degc + 1.f);
    const float s1 = dlog / AVG_LOG;
    const float s2 = AVG_LOG / dlog;
    _Float16* drow = ht + (long)ntid * 1664;
    const long base = (long)n * 512 + t * 128;

    for (int f2 = threadIdx.x; f2 < 832; f2 += blockDim.x) {
        int f = f2 << 1;
        float v0, v1;
        if (f < 128) {
            v0 = h[base + f];
            v1 = h[base + f + 1];
        } else {
            int g = f - 128;
            int grp = g / 512;
            int a = g - grp * 512;
            int comp = a >> 7;
            int c = a & 127;
            long off = base + c;
            float scale = (grp == 0) ? 1.f : (grp == 1 ? s1 : s2);
            if (comp == 0) {
                v0 = Ssum[off] * invd;
                v1 = Ssum[off + 1] * invd;
            } else if (comp == 1) {
                v0 = has ? fdec(Smax[off]) : 0.f;
                v1 = has ? fdec(Smax[off + 1]) : 0.f;
            } else if (comp == 2) {
                v0 = has ? fdec(Smin[off]) : 0.f;
                v1 = has ? fdec(Smin[off + 1]) : 0.f;
            } else {
                float m0 = Ssum[off] * invd, m1 = Ssum[off + 1] * invd;
                float var0 = Ssq[off] * invd - m0 * m0;
                float var1 = Ssq[off + 1] * invd - m1 * m1;
                var0 = var0 > 0.f ? var0 : 0.f;
                var1 = var1 > 0.f ? var1 : 0.f;
                v0 = sqrtf(var0 + 1e-5f);
                v1 = sqrtf(var1 + 1e-5f);
            }
            v0 *= scale; v1 *= scale;
        }
        v2h o = { (_Float16)v0, (_Float16)v1 };
        *(v2h*)&drow[f] = o;
    }
}

// ---------- K3b: posttrans  out = (ht @ postW[t] + b) * snorm  -> Xmid f16 ----------
// block = 128 thr (4 waves = 4 towers), 16 nodes.  K = 1664 = 52*32.
__global__ void k_posttrans(const _Float16* __restrict__ ht, const _Float16* __restrict__ postWt,
                            const float* __restrict__ postB, const float* __restrict__ snorm,
                            _Float16* __restrict__ Xmid) {
    const int nb = blockIdx.x * 16;
    const int t = threadIdx.x >> 5, lane = threadIdx.x & 31;
    v8f acc[8];
    #pragma unroll
    for (int i = 0; i < 8; ++i)
        #pragma unroll
        for (int j = 0; j < 8; ++j) acc[i][j] = 0.f;

    const _Float16* A0 = ht + ((long)nb * TT + t) * 1664;   // node row stride = 4*1664
    const _Float16* Bt0 = postWt + (long)t * 128 * 1664;
    const int arow = lane & 15, akb = (lane >> 4) << 3;
    for (int kk = 0; kk < 52; ++kk) {
        int k0 = kk * 32;
        v16h a = load_a_frag(A0 + k0, TT * 1664, lane);
        // prefetch this lane's A chunk two k-steps ahead (ht streams once from HBM)
        if (kk + 2 < 52)
            __builtin_prefetch(A0 + (kk + 2) * 32 + arow * (TT * 1664) + akb, 0, 1);
        v16h bfr[8];
        #pragma unroll
        for (int nt = 0; nt < 8; ++nt)
            bfr[nt] = load_b_frag(Bt0 + (long)nt * 16 * 1664 + k0, 1664, lane);
        #pragma unroll
        for (int nt = 0; nt < 8; ++nt)
            acc[nt] = __builtin_amdgcn_wmma_f32_16x16x32_f16(false, a, false, bfr[nt],
                                                             (short)0, acc[nt], false, false);
    }
    const int n15 = lane & 15, mOff = (lane >> 4) * 8;
    #pragma unroll
    for (int nt = 0; nt < 8; ++nt) {
        int col = nt * 16 + n15;
        float bias = postB[t * 128 + col];
        #pragma unroll
        for (int v = 0; v < 8; ++v) {
            int node = nb + mOff + v;
            float val = (acc[nt][v] + bias) * snorm[node];
            Xmid[(long)node * 512 + t * 128 + col] = (_Float16)val;
        }
    }
}

// ---------- K4: mix  out = h + leaky(Xmid @ mixW + b) ----------
// block = 128 thr (4 waves = 4 column groups of 128), 16 nodes.  K = 512 = 16*32.
__global__ void k_mix(const _Float16* __restrict__ Xmid, const _Float16* __restrict__ mixWt,
                      const float* __restrict__ mixB, const float* __restrict__ h,
                      float* __restrict__ out) {
    const int nb = blockIdx.x * 16;
    const int cg = threadIdx.x >> 5, lane = threadIdx.x & 31;
    v8f acc[8];
    #pragma unroll
    for (int i = 0; i < 8; ++i)
        #pragma unroll
        for (int j = 0; j < 8; ++j) acc[i][j] = 0.f;

    const _Float16* A0 = Xmid + (long)nb * 512;
    const _Float16* Bt0 = mixWt + (long)cg * 128 * 512;
    for (int kk = 0; kk < 16; ++kk) {
        int k0 = kk * 32;
        v16h a = load_a_frag(A0 + k0, 512, lane);
        v16h bfr[8];
        #pragma unroll
        for (int nt = 0; nt < 8; ++nt)
            bfr[nt] = load_b_frag(Bt0 + (long)nt * 16 * 512 + k0, 512, lane);
        #pragma unroll
        for (int nt = 0; nt < 8; ++nt)
            acc[nt] = __builtin_amdgcn_wmma_f32_16x16x32_f16(false, a, false, bfr[nt],
                                                             (short)0, acc[nt], false, false);
    }
    const int n15 = lane & 15, mOff = (lane >> 4) * 8;
    #pragma unroll
    for (int nt = 0; nt < 8; ++nt) {
        int col = cg * 128 + nt * 16 + n15;
        float bias = mixB[col];
        #pragma unroll
        for (int v = 0; v < 8; ++v) {
            int node = nb + mOff + v;
            float val = acc[nt][v] + bias;
            val = val > 0.f ? val : 0.01f * val;                // leaky_relu(0.01)
            out[(long)node * 512 + col] = h[(long)node * 512 + col] + val;
        }
    }
}

extern "C" void kernel_launch(void* const* d_in, const int* in_sizes, int n_in,
                              void* d_out, int out_size, void* d_ws, size_t ws_size,
                              hipStream_t stream) {
    const float* h     = (const float*)d_in[0];
    const float* e     = (const float*)d_in[1];
    const float* snorm = (const float*)d_in[2];
    const float* preW  = (const float*)d_in[3];
    const float* preB  = (const float*)d_in[4];
    const float* postW = (const float*)d_in[5];
    const float* postB = (const float*)d_in[6];
    const float* mixW  = (const float*)d_in[7];
    const float* mixB  = (const float*)d_in[8];
    const int*   src   = (const int*)d_in[9];
    const int*   dst   = (const int*)d_in[10];
    float* out = (float*)d_out;

    char* ws = (char*)d_ws;
    const size_t oDeg   = 0;
    const size_t oSum   = oDeg   + (size_t)NN * 4;                 // 32 KB
    const size_t oSq    = oSum   + (size_t)NN * 512 * 4;           // 16.8 MB each
    const size_t oMax   = oSq    + (size_t)NN * 512 * 4;
    const size_t oMin   = oMax   + (size_t)NN * 512 * 4;
    const size_t oPreW  = oMin   + (size_t)NN * 512 * 4;
    const size_t oPostW = oPreW  + (size_t)TT * 128 * 288 * 2;     // 294 KB
    const size_t oMixW  = oPostW + (size_t)TT * 128 * 1664 * 2;    // 1.7 MB
    const size_t oHt    = oMixW  + (size_t)512 * 512 * 2;          // 512 KB
    const size_t oX     = oHt    + (size_t)NN * TT * 1664 * 2;     // 109 MB

    unsigned*  dDeg   = (unsigned*)(ws + oDeg);
    float*     dSum   = (float*)(ws + oSum);
    float*     dSq    = (float*)(ws + oSq);
    unsigned*  dMax   = (unsigned*)(ws + oMax);
    unsigned*  dMin   = (unsigned*)(ws + oMin);
    _Float16*  dPreW  = (_Float16*)(ws + oPreW);
    _Float16*  dPostW = (_Float16*)(ws + oPostW);
    _Float16*  dMixW  = (_Float16*)(ws + oMixW);
    _Float16*  dHt    = (_Float16*)(ws + oHt);
    _Float16*  dX     = (_Float16*)(ws + oX);

    // weight convert+transpose to f16 Bt[n][k]
    {
        dim3 g((288 * 128 + 255) / 256, TT);
        k_w2h<<<g, 256, 0, stream>>>(preW, dPreW, 288, 128);
    }
    {
        dim3 g((1664 * 128 + 255) / 256, TT);
        k_w2h<<<g, 256, 0, stream>>>(postW, dPostW, 1664, 128);
    }
    {
        dim3 g((512 * 512 + 255) / 256, 1);
        k_w2h<<<g, 256, 0, stream>>>(mixW, dMixW, 512, 512);
    }
    // clear accumulators
    k_init<<<(NN * 512 + 255) / 256, 256, 0, stream>>>(dSum, dSq, dMax, dMin, dDeg);
    // degrees
    k_deg<<<(EE + 255) / 256, 256, 0, stream>>>(dst, dDeg);
    // pretrans GEMM + scatter aggregation
    k_pretrans<<<EE / 16, 128, 0, stream>>>(h, e, preB, src, dst, dPreW,
                                            dSum, dSq, dMax, dMin);
    // build ht
    k_build_ht<<<NN * TT, 256, 0, stream>>>(h, dSum, dSq, dMax, dMin, dDeg, dHt);
    // posttrans GEMM
    k_posttrans<<<NN / 16, 128, 0, stream>>>(dHt, dPostW, postB, snorm, dX);
    // mix GEMM + leaky relu + residual
    k_mix<<<NN / 16, 128, 0, stream>>>(dX, dMixW, mixB, h, out);
}
